// BatteryMoEFlattenIntraCycleMoELayer_17154099381009
// MI455X (gfx1250) — matrix operand order repaired and checked by reading, not compiled, not run on hardware
//
#include <hip/hip_runtime.h>
#include <hip/hip_bf16.h>

#define B_    512
#define L_    128
#define FEAT_ 900
#define D_    512
#define E_    8
#define BK    32
#define KFULL (FEAT_ / BK)               // 28 full K-steps
#define KTAIL (FEAT_ - KFULL * BK)       // 4 remaining K values

typedef __attribute__((ext_vector_type(16))) _Float16 v16h;
typedef __attribute__((ext_vector_type(8)))  _Float16 v8h;
typedef __attribute__((ext_vector_type(4)))  _Float16 v4h;
typedef __attribute__((ext_vector_type(8)))  float    v8f;
typedef __attribute__((ext_vector_type(4)))  float    f4v;

// ---------------- Kernel 1: gates (softmax -> mask -> top2 -> renorm) ----------------
__global__ void moe_gates_kernel(const float* __restrict__ logits,
                                 const int*   __restrict__ masks,
                                 float* __restrict__ g0o, float* __restrict__ g1o,
                                 int*   __restrict__ e0o, int*   __restrict__ e1o) {
  int b = blockIdx.x * blockDim.x + threadIdx.x;
  if (b >= B_) return;
  float l[E_];
  float mx = -1e30f;
#pragma unroll
  for (int e = 0; e < E_; ++e) { l[e] = logits[b * E_ + e]; mx = fmaxf(mx, l[e]); }
  float s = 0.f;
#pragma unroll
  for (int e = 0; e < E_; ++e) { l[e] = __expf(l[e] - mx); s += l[e]; }
  float inv = 1.f / s;
#pragma unroll
  for (int e = 0; e < E_; ++e) {
    float g = l[e] * inv;
    if (masks[b * E_ + e] != 1) g = 0.f;
    l[e] = g;
  }
  int i0 = 0;
#pragma unroll
  for (int e = 1; e < E_; ++e) if (l[e] > l[i0]) i0 = e;
  int i1 = (i0 == 0) ? 1 : 0;
#pragma unroll
  for (int e = 0; e < E_; ++e) if (e != i0 && l[e] > l[i1]) i1 = e;
  float denom = l[i0] + l[i1] + 1e-9f;
  g0o[b] = l[i0] / denom;
  g1o[b] = l[i1] / denom;
  e0o[b] = i0;
  e1o[b] = i1;
}

// ---------------- Kernel 2: per-batch GEMM with gate-combined expert weights --------
// grid = (D_/128, B_), block = 256 threads = 8 wave32s.
// Block tile: M=128 (all of L), N=128, K-step=32. Wave tile: 64x32 = 4x2 WMMA 16x16x32 f16.
__global__ __launch_bounds__(256) void moe_wmma_gemm(
    const float* __restrict__ X,      // [B_, L_, FEAT_]
    const float* __restrict__ W,      // [E_, FEAT_, D_]
    const float* __restrict__ bias,   // [E_, D_]
    const float* __restrict__ g0a, const float* __restrict__ g1a,
    const int*   __restrict__ e0a, const int*   __restrict__ e1a,
    __hip_bfloat16* __restrict__ out) // [B_, L_, D_]
{
  __shared__ __align__(16) _Float16 sA[L_][40];   // [m][k], stride 40 f16 (=80B)
  __shared__ __align__(16) _Float16 sB[128][40];  // [n][k], stride 40 f16

  const int b  = blockIdx.y;
  const int n0 = blockIdx.x * 128;

  const int e0 = e0a[b];
  const int e1 = e1a[b];
  const float g0 = g0a[b];
  const float g1 = g1a[b];

  const float* A  = X + (size_t)b * (L_ * FEAT_);
  const float* W0 = W + (size_t)e0 * (FEAT_ * D_);
  const float* W1 = W + (size_t)e1 * (FEAT_ * D_);

  const int tid  = threadIdx.x;
  // A loader: row ra (0..127), 16-float half ha (0/1)
  const int ra = tid >> 1;
  const int ha = tid & 1;
  // B loader: 4x4 (k x d) sub-tile per thread: dt = (tid&31)*4, kt = (tid>>5)*4
  const int dt = (tid & 31) * 4;
  const int kt = (tid >> 5) * 4;

  // wave geometry: 2 wave-rows x 4 wave-cols
  const int wave = tid >> 5;
  const int lane = tid & 31;
  const int wm   = wave >> 2;      // 0..1 -> rows [wm*64, wm*64+64)
  const int wn   = wave & 3;       // 0..3 -> cols [wn*32, wn*32+32)
  const int lh   = lane & 15;
  const int lsel = lane >> 4;      // 0 = lanes 0-15, 1 = lanes 16-31

  v8f acc[4][2];
#pragma unroll
  for (int mi = 0; mi < 4; ++mi)
#pragma unroll
    for (int ni = 0; ni < 2; ++ni)
      acc[mi][ni] = (v8f){0.f, 0.f, 0.f, 0.f, 0.f, 0.f, 0.f, 0.f};

  // fragment loads + 8 WMMAs from the current LDS tile
  auto compute_tile = [&]() {
    // A 16x32: lane<16 holds M=lh, K 0-7 & 16-23; lane>=16 holds K 8-15 & 24-31
    v16h afrag[4];
#pragma unroll
    for (int mi = 0; mi < 4; ++mi) {
      const int row = wm * 64 + mi * 16 + lh;
      const int c0  = lsel * 8;
      v8h lo = *(const v8h*)&sA[row][c0];
      v8h hi = *(const v8h*)&sA[row][c0 + 16];
      afrag[mi] = __builtin_shufflevector(lo, hi, 0, 1, 2, 3, 4, 5, 6, 7,
                                          8, 9, 10, 11, 12, 13, 14, 15);
    }
    // B 32x16: lane<16 holds N=lh, K 0-7 & 16-23; lane>=16 holds K 8-15 & 24-31
    v16h bfrag[2];
#pragma unroll
    for (int ni = 0; ni < 2; ++ni) {
      const int col = wn * 32 + ni * 16 + lh;
      const int c0  = lsel * 8;
      v8h lo = *(const v8h*)&sB[col][c0];
      v8h hi = *(const v8h*)&sB[col][c0 + 16];
      bfrag[ni] = __builtin_shufflevector(lo, hi, 0, 1, 2, 3, 4, 5, 6, 7,
                                          8, 9, 10, 11, 12, 13, 14, 15);
    }
#pragma unroll
    for (int mi = 0; mi < 4; ++mi)
#pragma unroll
      for (int ni = 0; ni < 2; ++ni)
        acc[mi][ni] = __builtin_amdgcn_wmma_f32_16x16x32_f16(
            false, afrag[mi], false, bfrag[ni],
            (short)0, acc[mi][ni], false, false);
  };

  // stores aR (16 floats, A row segment) and cb[kr][dd] (4x4 combined W) into LDS
  auto store_tile = [&](const float* aR, const float (*cb)[4]) {
    v8h a0, a1;
#pragma unroll
    for (int j = 0; j < 8; ++j) {
      a0[j] = (_Float16)aR[j];
      a1[j] = (_Float16)aR[j + 8];
    }
    *(v8h*)&sA[ra][ha * 16]     = a0;
    *(v8h*)&sA[ra][ha * 16 + 8] = a1;
#pragma unroll
    for (int dd = 0; dd < 4; ++dd) {
      v4h h;
#pragma unroll
      for (int kr = 0; kr < 4; ++kr) h[kr] = (_Float16)cb[kr][dd];
      *(v4h*)&sB[dt + dd][kt] = h;     // 8B-aligned: kt % 4 == 0, row stride 80B
    }
  };

  // ---------------- main loop: 28 full, unguarded K-steps ----------------
  for (int kk = 0; kk < KFULL; ++kk) {
    const int k0 = kk * BK;

    float aR[16];
    const float* arow = A + (size_t)ra * FEAT_ + k0 + ha * 16;
#pragma unroll
    for (int j = 0; j < 16; j += 4) {
      f4v t = *(const f4v*)(arow + j);
      aR[j] = t[0]; aR[j + 1] = t[1]; aR[j + 2] = t[2]; aR[j + 3] = t[3];
    }

    float cb[4][4];
    const float* w0p = W0 + (size_t)(k0 + kt) * D_ + n0 + dt;
    const float* w1p = W1 + (size_t)(k0 + kt) * D_ + n0 + dt;
#pragma unroll
    for (int kr = 0; kr < 4; ++kr) {
      f4v t0 = *(const f4v*)(w0p + (size_t)kr * D_);
      f4v t1 = *(const f4v*)(w1p + (size_t)kr * D_);
#pragma unroll
      for (int dd = 0; dd < 4; ++dd)
        cb[kr][dd] = fmaf(g1, t1[dd], g0 * t0[dd]);
    }

    __syncthreads();              // previous compute done reading LDS
    store_tile(aR, cb);
    __syncthreads();              // tile ready
    compute_tile();
  }

  // ---------------- tail: k0 = 896, 4 valid K values, zero-padded ----------------
  {
    const int k0 = KFULL * BK;

    float aR[16];
    const float* arow = A + (size_t)ra * FEAT_ + k0 + ha * 16;
#pragma unroll
    for (int j = 0; j < 16; ++j) {
      int kidx = k0 + ha * 16 + j;
      aR[j] = (kidx < FEAT_) ? arow[j] : 0.f;
    }

    float cb[4][4];
    const float* w0p = W0 + (size_t)(k0 + kt) * D_ + n0 + dt;
    const float* w1p = W1 + (size_t)(k0 + kt) * D_ + n0 + dt;
#pragma unroll
    for (int kr = 0; kr < 4; ++kr) {
      int krow = k0 + kt + kr;
      if (krow < FEAT_) {
        f4v t0 = *(const f4v*)(w0p + (size_t)kr * D_);
        f4v t1 = *(const f4v*)(w1p + (size_t)kr * D_);
#pragma unroll
        for (int dd = 0; dd < 4; ++dd)
          cb[kr][dd] = fmaf(g1, t1[dd], g0 * t0[dd]);
      } else {
#pragma unroll
        for (int dd = 0; dd < 4; ++dd) cb[kr][dd] = 0.f;
      }
    }

    __syncthreads();
    store_tile(aR, cb);
    __syncthreads();
    compute_tile();
  }

  // ---------------- epilogue: add gate-combined bias, store bf16 ----------------
#pragma unroll
  for (int ni = 0; ni < 2; ++ni) {
    const int col = n0 + wn * 32 + ni * 16 + lh;
    const float bc = fmaf(g1, bias[e1 * D_ + col], g0 * bias[e0 * D_ + col]);
#pragma unroll
    for (int mi = 0; mi < 4; ++mi) {
      const int mbase = wm * 64 + mi * 16 + lsel * 8;   // C: VGPR r -> row mbase+r
#pragma unroll
      for (int r = 0; r < 8; ++r) {
        float v = acc[mi][ni][r] + bc;
        out[((size_t)b * L_ + (mbase + r)) * D_ + col] = __float2bfloat16(v);
      }
    }
  }
}

extern "C" void kernel_launch(void* const* d_in, const int* in_sizes, int n_in,
                              void* d_out, int out_size, void* d_ws, size_t ws_size,
                              hipStream_t stream) {
  (void)in_sizes; (void)n_in; (void)out_size; (void)ws_size;
  const float* X      = (const float*)d_in[0];  // cycle_curve_data [512,128,3,300]
  const float* logits = (const float*)d_in[1];  // [512,8]
  const int*   masks  = (const int*)d_in[2];    // [512,8]
  const float* W      = (const float*)d_in[3];  // [8,900,512]
  const float* bias   = (const float*)d_in[4];  // [8,512]
  __hip_bfloat16* out = (__hip_bfloat16*)d_out; // [512,128,512]

  float* g0 = (float*)d_ws;
  float* g1 = g0 + B_;
  int*   e0 = (int*)(g1 + B_);
  int*   e1 = e0 + B_;

  moe_gates_kernel<<<dim3((B_ + 255) / 256), dim3(256), 0, stream>>>(
      logits, masks, g0, g1, e0, e1);

  moe_wmma_gemm<<<dim3(D_ / 128, B_), dim3(256), 0, stream>>>(
      X, W, bias, g0, g1, e0, e1, out);
}
